// LowRankRateRNN_84868553769011
// MI455X (gfx1250) — matrix operand order repaired and checked by reading, not compiled, not run on hardware
//
#include <hip/hip_runtime.h>

// ---------------- problem constants (match reference setup_inputs) -------
constexpr int   BATCH   = 64;
constexpr int   TSTEPS  = 1024;
constexpr int   IDIM    = 128;
constexpr int   HDIM    = 512;
constexpr int   ODIM    = 512;
constexpr float ALPHA_C = 0.1f;

// ---------------- WMMA fragment types ------------------------------------
typedef __attribute__((ext_vector_type(16))) __bf16 v16bf;
typedef __attribute__((ext_vector_type(8)))  float  v8f;

union Frag16 { v16bf v; uint4 q[2]; };
union FragC  { v8f  v; float f[8]; };

// pointer types for the gfx1250 async-to-LDS builtin (b128 = int4 payload)
typedef int v4i __attribute__((vector_size(16)));
typedef __attribute__((address_space(1))) v4i* gptr_b128;
typedef __attribute__((address_space(3))) v4i* lptr_b128;

__device__ __forceinline__ unsigned short f32_to_bf16(float f) {
  unsigned int u = __float_as_uint(f);
  u += 0x7FFFu + ((u >> 16) & 1u);      // round-to-nearest-even
  return (unsigned short)(u >> 16);
}

// ---------------- f32 -> bf16 convert (grid-stride) ----------------------
__global__ void f32_to_bf16_kernel(const float* __restrict__ in,
                                   unsigned short* __restrict__ out, int n) {
  int i = blockIdx.x * blockDim.x + threadIdx.x;
  const int stride = gridDim.x * blockDim.x;
  for (; i < n; i += stride) out[i] = f32_to_bf16(in[i]);
}

// ---------------- bf16 WMMA GEMM: C(M,N)f32 = A(M,K) * Bt(N,K)^T + bias --
// Block: 256 threads = 8 waves, tile 128(M) x 64(N), K templated (128/512).
//
// The 64xK B strip is shared by all 8 waves -> staged once into LDS via
// GLOBAL_LOAD_ASYNC_TO_LDS_B128 (ASYNCcnt path; plain-load fallback if the
// builtin is unavailable). LDS rows padded +8 elems so each half-wave's 16
// lanes read disjoint bank quads (conflict-free ds_load_b128).
//
// A fragments are per-wave global b128 loads, double-buffered one K-chunk
// ahead so VMEM latency hides behind the WMMAs instead of s_wait stalls.
template <int K>
__global__ void __launch_bounds__(256) gemm_bf16_wmma_kernel(
    const unsigned short* __restrict__ A,   // (M,K) bf16 row-major
    const unsigned short* __restrict__ Bt,  // (N,K) bf16 row-major == B^T
    const float* __restrict__ bias,         // (N)
    float* __restrict__ C,                  // (M,N) f32
    int M, int N)
{
  constexpr int KP = K + 8;                 // padded LDS row stride (elems)
  __shared__ unsigned short btile[64 * KP];

  const int lane = threadIdx.x & 31;
  const int wave = threadIdx.x >> 5;
  const int half = lane >> 4;               // 0: lanes 0-15, 1: lanes 16-31
  const int l16  = lane & 15;
  const int m0   = blockIdx.y * 128 + wave * 16;
  const int nblk = blockIdx.x * 64;

  // ---- stage 64 x K B-tile into LDS (16B segments) -----------------------
  {
    constexpr int SEGS_PER_ROW = K / 8;
    constexpr int TOTAL = 64 * SEGS_PER_ROW;
    const int tid = threadIdx.x;
#if __has_builtin(__builtin_amdgcn_global_load_async_to_lds_b128)
#pragma unroll
    for (int c0 = 0; c0 < TOTAL; c0 += 256) {
      const int c = c0 + tid;
      const int r = c / SEGS_PER_ROW;
      const int s = c % SEGS_PER_ROW;
      const unsigned short* src = Bt + (size_t)(nblk + r) * K + s * 8;
      unsigned short* dst = &btile[r * KP + s * 8];
      __builtin_amdgcn_global_load_async_to_lds_b128(
          (gptr_b128)src, (lptr_b128)dst, 0, 0);
    }
#if __has_builtin(__builtin_amdgcn_s_wait_asynccnt)
    __builtin_amdgcn_s_wait_asynccnt(0);
#else
    asm volatile("s_wait_asynccnt 0x0" ::: "memory");
#endif
#else
#pragma unroll
    for (int c0 = 0; c0 < TOTAL; c0 += 256) {
      const int c = c0 + tid;
      const int r = c / SEGS_PER_ROW;
      const int s = c % SEGS_PER_ROW;
      const uint4* src =
          reinterpret_cast<const uint4*>(Bt + (size_t)(nblk + r) * K + s * 8);
      *reinterpret_cast<uint4*>(&btile[r * KP + s * 8]) = *src;
    }
#endif
    __syncthreads();
  }

  FragC c[4];
#pragma unroll
  for (int nt = 0; nt < 4; ++nt) {
    const float bv = bias[nblk + nt * 16 + l16];
#pragma unroll
    for (int r = 0; r < 8; ++r) c[nt].f[r] = bv;
  }

  const size_t arowOff = (size_t)(m0 + l16) * K;
  constexpr int NCHUNK = K / 32;

  // prime A double-buffer (chunk 0)
  Frag16 a_cur, a_nxt;
  {
    const uint4* pa = reinterpret_cast<const uint4*>(A + arowOff + half * 8);
    a_cur.q[0] = pa[0];             // K = base+0..7   (pairs in VGPR0-3)
    a_cur.q[1] = pa[2];             // K = base+16..23 (pairs in VGPR4-7)
  }

#pragma unroll
  for (int kc = 0; kc < NCHUNK; ++kc) {
    const int k0 = kc * 32;
    if (kc + 1 < NCHUNK) {          // issue next chunk's A loads early
      const uint4* pa =
          reinterpret_cast<const uint4*>(A + arowOff + k0 + 32 + half * 8);
      a_nxt.q[0] = pa[0];
      a_nxt.q[1] = pa[2];
    }
#pragma unroll
    for (int nt = 0; nt < 4; ++nt) {
      const int rloc = nt * 16 + l16;      // column within 64-wide strip
      const uint4* pb =
          reinterpret_cast<const uint4*>(&btile[rloc * KP + k0 + half * 8]);
      Frag16 bfrag;
      bfrag.q[0] = pb[0];
      bfrag.q[1] = pb[2];
      c[nt].v = __builtin_amdgcn_wmma_f32_16x16x32_bf16(
          false, a_cur.v, false, bfrag.v, (short)0, c[nt].v, false, false);
    }
    a_cur = a_nxt;
  }

  // C layout: VGPR r -> (m0+r, n) lanes 0-15 ; (m0+8+r, n) lanes 16-31
  const int rbase = m0 + half * 8;
#pragma unroll
  for (int nt = 0; nt < 4; ++nt) {
    const int n = nblk + nt * 16 + l16;
#pragma unroll
    for (int r = 0; r < 8; ++r)
      C[(size_t)(rbase + r) * N + n] = c[nt].f[r];
  }
}

// ---------------- sequential low-rank recurrence -------------------------
// One wave32 per batch element (4 waves / 128-thread block). Each lane owns
// 16 contiguous hidden units -> rank-2 projection reduces with __shfl_xor
// only (no LDS, no barriers). xproj loads are software-pipelined one step
// ahead; h stays in registers for all 1024 steps. Writes h (f32) in place
// over xproj in d_out and a bf16 copy for the output GEMM.
__global__ void __launch_bounds__(128) lowrank_scan_kernel(
    const float* __restrict__ U, const float* __restrict__ V,
    const float* __restrict__ h0,
    float* __restrict__ hall,            // (B,T,H) f32: xproj in, h out
    unsigned short* __restrict__ hbf)    // (B,T,H) bf16 out
{
  const int lane = threadIdx.x & 31;
  const int wave = threadIdx.x >> 5;
  const int b    = blockIdx.x * 4 + wave;
  const int j0   = lane * 16;

  float h[16], u0[16], u1[16], v0[16], v1[16];
  {
    const float4* U4 = reinterpret_cast<const float4*>(U + (size_t)j0 * 2);
    const float4* V4 = reinterpret_cast<const float4*>(V + (size_t)j0 * 2);
#pragma unroll
    for (int i = 0; i < 8; ++i) {   // (H,2) row-major: interleaved r=0,1
      float4 tu = U4[i];
      u0[2*i] = tu.x; u1[2*i] = tu.y; u0[2*i+1] = tu.z; u1[2*i+1] = tu.w;
      float4 tv = V4[i];
      v0[2*i] = tv.x; v1[2*i] = tv.y; v0[2*i+1] = tv.z; v1[2*i+1] = tv.w;
    }
    const float4* H4 =
        reinterpret_cast<const float4*>(h0 + (size_t)b * HDIM + j0);
#pragma unroll
    for (int i = 0; i < 4; ++i) {
      float4 t = H4[i];
      h[4*i] = t.x; h[4*i+1] = t.y; h[4*i+2] = t.z; h[4*i+3] = t.w;
    }
  }

  size_t base = (size_t)b * TSTEPS * HDIM + j0;
  float xp[16];
#pragma unroll
  for (int i = 0; i < 4; ++i) {
    float4 t = reinterpret_cast<const float4*>(hall + base)[i];
    xp[4*i] = t.x; xp[4*i+1] = t.y; xp[4*i+2] = t.z; xp[4*i+3] = t.w;
  }

  const float scale = 1.0f / (float)HDIM;
  for (int t = 0; t < TSTEPS; ++t) {
    float xpn[16] = {};
    const size_t nbase = base + HDIM;
    if (t + 1 < TSTEPS) {            // prefetch next step's xproj
#pragma unroll
      for (int i = 0; i < 4; ++i) {
        float4 q = reinterpret_cast<const float4*>(hall + nbase)[i];
        xpn[4*i] = q.x; xpn[4*i+1] = q.y; xpn[4*i+2] = q.z; xpn[4*i+3] = q.w;
      }
    }

    float p0 = 0.f, p1 = 0.f;        // partial h @ V  (rank 2)
#pragma unroll
    for (int i = 0; i < 16; ++i) {
      p0 = fmaf(h[i], v0[i], p0);
      p1 = fmaf(h[i], v1[i], p1);
    }
#pragma unroll
    for (int off = 16; off >= 1; off >>= 1) {   // wave32 all-reduce
      p0 += __shfl_xor(p0, off, 32);
      p1 += __shfl_xor(p1, off, 32);
    }

#pragma unroll
    for (int i = 0; i < 16; ++i) {
      float pre = fmaf(fmaf(u0[i], p0, u1[i] * p1), scale, xp[i]);
      h[i] = fmaf(1.0f - ALPHA_C, h[i], ALPHA_C * tanhf(pre));
    }

    float4* D4 = reinterpret_cast<float4*>(hall + base);
#pragma unroll
    for (int i = 0; i < 4; ++i)
      D4[i] = make_float4(h[4*i], h[4*i+1], h[4*i+2], h[4*i+3]);

    unsigned int pk[8];
#pragma unroll
    for (int i = 0; i < 8; ++i)
      pk[i] = (unsigned int)f32_to_bf16(h[2*i]) |
              ((unsigned int)f32_to_bf16(h[2*i+1]) << 16);
    uint4* Bf = reinterpret_cast<uint4*>(hbf + base);
    Bf[0] = make_uint4(pk[0], pk[1], pk[2], pk[3]);
    Bf[1] = make_uint4(pk[4], pk[5], pk[6], pk[7]);

#pragma unroll
    for (int i = 0; i < 16; ++i) xp[i] = xpn[i];
    base = nbase;
  }
}

// ---------------- host-side launch ---------------------------------------
extern "C" void kernel_launch(void* const* d_in, const int* in_sizes, int n_in,
                              void* d_out, int out_size, void* d_ws,
                              size_t ws_size, hipStream_t stream) {
  const float* x   = (const float*)d_in[0];
  const float* h0  = (const float*)d_in[1];
  const float* Wxh = (const float*)d_in[2];  // (H,I)
  const float* bxh = (const float*)d_in[3];
  const float* U   = (const float*)d_in[4];  // (H,2)
  const float* V   = (const float*)d_in[5];  // (H,2)
  const float* Why = (const float*)d_in[6];  // (O,H)
  const float* bhy = (const float*)d_in[7];

  const size_t BT = (size_t)BATCH * TSTEPS;          // 65536
  float* y    = (float*)d_out;                       // (B,T,O)
  float* hall = y + BT * ODIM;                       // (B,T,H) second output

  // workspace: bf16 copies (x: 16.8MB, Wxh: 128KB, Why: 512KB, h: 67MB)
  unsigned short* x_bf   = (unsigned short*)d_ws;
  unsigned short* Wxh_bf = x_bf   + BT * IDIM;
  unsigned short* Why_bf = Wxh_bf + (size_t)HDIM * IDIM;
  unsigned short* h_bf   = Why_bf + (size_t)ODIM * HDIM;

  f32_to_bf16_kernel<<<2048, 256, 0, stream>>>(x, x_bf, (int)(BT * IDIM));
  f32_to_bf16_kernel<<<64, 256, 0, stream>>>(Wxh, Wxh_bf, HDIM * IDIM);
  f32_to_bf16_kernel<<<256, 256, 0, stream>>>(Why, Why_bf, ODIM * HDIM);

  // xproj = x @ Wxh^T + bxh   -> written into hall (consumed in-place)
  gemm_bf16_wmma_kernel<IDIM>
      <<<dim3(HDIM / 64, (unsigned)(BT / 128)), 256, 0, stream>>>(
          x_bf, Wxh_bf, bxh, hall, (int)BT, HDIM);

  // sequential low-rank recurrence (wave-per-batch, shfl-only reduction)
  lowrank_scan_kernel<<<BATCH / 4, 128, 0, stream>>>(U, V, h0, hall, h_bf);

  // y = h_all @ Why^T + bhy
  gemm_bf16_wmma_kernel<HDIM>
      <<<dim3(ODIM / 64, (unsigned)(BT / 128)), 256, 0, stream>>>(
          h_bf, Why_bf, bhy, y, (int)BT, ODIM);

  (void)in_sizes; (void)n_in; (void)out_size; (void)ws_size;
}
